// TimeAwareEncoderBlock_58274116272259
// MI455X (gfx1250) — matrix operand order repaired and checked by reading, not compiled, not run on hardware
//
#include <hip/hip_runtime.h>
#include <math.h>

// ---- problem dims ----
#define B_   2
#define S_   2048
#define D_   1024
#define H_   16
#define G_   4
#define HD_  64
#define E_   8
#define FF_  4096
#define NT   (B_*S_)     // 4096 tokens

typedef __bf16 bf16;
typedef bf16  v16bf __attribute__((ext_vector_type(16)));
typedef float v8f   __attribute__((ext_vector_type(8)));
typedef unsigned int v8u __attribute__((ext_vector_type(8)));

__device__ __forceinline__ unsigned short f2bf(float f) {
  unsigned u = __float_as_uint(f);
  u += 0x7FFFu + ((u >> 16) & 1u);           // round-to-nearest-even
  return (unsigned short)(u >> 16);
}

// ---- gfx1250 async global->LDS copy (ISA 10.x / CDNA5 ISA) ----
// LDS byte address = low 32 bits of the generic pointer (LDS aperture mapping).
__device__ __forceinline__ void async_g2l_b64(void* lds, const void* g) {
  unsigned l = (unsigned)(unsigned long long)lds;
  asm volatile("global_load_async_to_lds_b64 %0, %1, off" :: "v"(l), "v"(g) : "memory");
}
__device__ __forceinline__ void wait_async0() {
  asm volatile("s_wait_asynccnt 0" ::: "memory");
}

// ---- WMMA fragment loaders (bf16, 16x16x32), per ISA 7.12.2 layouts ----
// A: 16(M) x 32(K) row-major tile, row stride ld (elements, even).
__device__ __forceinline__ v16bf frag_a16(const unsigned short* p, int ld) {
  int lane = threadIdx.x & 31;
  int m = lane & 15, half = lane >> 4;
  const unsigned short* row = p + (long)m * ld;
  v8u r;
#pragma unroll
  for (int v = 0; v < 8; ++v) {
    int k = ((v & 4) << 2) + (half << 3) + ((v & 3) << 1); // (v>=4?16:0)+half*8+(v&3)*2
    r[v] = *(const unsigned*)(row + k);
  }
  return __builtin_bit_cast(v16bf, r);
}
// B: 32(K) x 16(N) row-major tile, row stride ld (elements, even). lane%16 = N column, K in same order as A.
__device__ __forceinline__ v16bf frag_b16(const unsigned short* p, int ld) {
  int lane = threadIdx.x & 31;
  int n = lane & 15, half = lane >> 4;
  v8u r;
#pragma unroll
  for (int v = 0; v < 8; ++v) {
    int k = ((v & 4) << 2) + (half << 3) + ((v & 3) << 1);
    r[v] = (unsigned)p[(long)k * ld + n] | ((unsigned)p[(long)(k + 1) * ld + n] << 16);
  }
  return __builtin_bit_cast(v16bf, r);
}
#define WMMA_BF16(acc, a, b) \
  acc = __builtin_amdgcn_wmma_f32_16x16x32_bf16(false, a, false, b, (short)0, acc, false, false)

// ===================== small utility kernels =====================
__global__ void k_cumsum(const float* deltas, float* t) {
  if (threadIdx.x == 0) {
    int b = blockIdx.x;
    float acc = 0.f;
    for (int s = 0; s < S_; ++s) { acc += deltas[b * S_ + s]; t[b * S_ + s] = acc; }
  }
}

__global__ void k_convert(const float* __restrict__ src, unsigned short* __restrict__ dst,
                          int rows, int cols, int dstStride, int colOff) {
  long total = (long)rows * cols;
  for (long i = (long)blockIdx.x * blockDim.x + threadIdx.x; i < total;
       i += (long)gridDim.x * blockDim.x) {
    long r = i / cols; int c = (int)(i - r * cols);
    dst[r * dstStride + colOff + c] = f2bf(src[i]);
  }
}

__device__ __forceinline__ float blockSum256(float v, float* sm) {
#pragma unroll
  for (int o = 16; o; o >>= 1) v += __shfl_xor(v, o, 32);
  if ((threadIdx.x & 31) == 0) sm[threadIdx.x >> 5] = v;
  __syncthreads();
  float r = 0.f;
#pragma unroll
  for (int i = 0; i < 8; ++i) r += sm[i];
  __syncthreads();
  return r;
}

__global__ __launch_bounds__(256) void k_ln1(const float* __restrict__ x,
                                             const float* __restrict__ sc,
                                             const float* __restrict__ bi,
                                             unsigned short* __restrict__ xn) {
  __shared__ float sm[8];
  long base = (long)blockIdx.x * D_;
  float s = 0.f;
  for (int d = threadIdx.x; d < D_; d += 256) s += x[base + d];
  float mu = blockSum256(s, sm) * (1.f / D_);
  float vs = 0.f;
  for (int d = threadIdx.x; d < D_; d += 256) { float e = x[base + d] - mu; vs += e * e; }
  float var = blockSum256(vs, sm) * (1.f / D_);
  float inv = rsqrtf(var + 1e-6f);
  for (int d = threadIdx.x; d < D_; d += 256)
    xn[base + d] = f2bf((x[base + d] - mu) * inv * sc[d] + bi[d]);
}

// ===================== QKV projection GEMM =====================
// C[4096,1536] = xn_bf[4096,1024] x Wqkv_bf[1024,1536]; scatter into q / kT / v (bf16)
__device__ __forceinline__ void store_qkv(const v8f& acc, int mrow, int ncol,
    const float* bq, const float* bk, const float* bv,
    unsigned short* qb, unsigned short* kTb, unsigned short* vb) {
  int lane = threadIdx.x & 31;
  int n = ncol + (lane & 15);
  int half = lane >> 4;
  float bias = (n < 1024) ? bq[n] : (n < 1280) ? bk[n - 1024] : bv[n - 1280];
#pragma unroll
  for (int j = 0; j < 8; ++j) {
    int m = mrow + j + half * 8;
    int b = m / S_, s = m - b * S_;
    unsigned short v16 = f2bf(acc[j] + bias);
    if (n < 1024) {
      int h = n >> 6, d = n & 63;
      qb[(((long)(b * H_ + h) * S_) + s) * HD_ + d] = v16;
    } else if (n < 1280) {
      int g = (n - 1024) >> 6, d = (n - 1024) & 63;
      kTb[(((long)(b * G_ + g) * HD_) + d) * S_ + s] = v16;      // k transposed [g][d][s]
    } else {
      int g = (n - 1280) >> 6, d = (n - 1280) & 63;
      vb[(((long)(b * G_ + g) * S_) + s) * HD_ + d] = v16;       // v [g][s][d]
    }
  }
}

__global__ __launch_bounds__(256) void k_gemm_qkv(
    const unsigned short* __restrict__ A, const unsigned short* __restrict__ Wqkv,
    const float* __restrict__ bq, const float* __restrict__ bk, const float* __restrict__ bv,
    unsigned short* qb, unsigned short* kTb, unsigned short* vb) {
  __shared__ unsigned short As[2][64][36];   // 36-elem stride: 8B aligned rows, bank-clean
  int m0 = blockIdx.x * 64, n0 = blockIdx.y * 64;
  int wave = threadIdx.x >> 5;
  int wm = wave & 3, wn = wave >> 2;        // 4x2 wave grid: 16 rows x 32 cols each
  int wn0 = n0 + wn * 32;
  v8f acc0 = {}, acc1 = {};
  auto stage = [&](int buf, int k0) {        // async global->LDS, 64 rows x 32 elems
    for (int i = threadIdx.x; i < 512; i += 256) {
      int r = i >> 3, c = (i & 7) << 2;
      async_g2l_b64(&As[buf][r][c], A + (long)(m0 + r) * D_ + k0 + c);
    }
  };
  stage(0, 0);
  wait_async0();
  __syncthreads();
  int buf = 0;
  for (int k0 = 0; k0 < D_; k0 += 32) {
    if (k0 + 32 < D_) stage(buf ^ 1, k0 + 32);        // prefetch next tile async
    __builtin_prefetch(Wqkv + (long)(k0 + 32) * 1536 + wn0, 0, 1);
    v16bf af = frag_a16(&As[buf][wm * 16][0], 36);
    v16bf b0 = frag_b16(Wqkv + (long)k0 * 1536 + wn0, 1536);
    v16bf b1 = frag_b16(Wqkv + (long)k0 * 1536 + wn0 + 16, 1536);
    WMMA_BF16(acc0, af, b0);
    WMMA_BF16(acc1, af, b1);
    wait_async0();
    __syncthreads();
    buf ^= 1;
  }
  store_qkv(acc0, m0 + wm * 16, wn0,      bq, bk, bv, qb, kTb, vb);
  store_qkv(acc1, m0 + wm * 16, wn0 + 16, bq, bk, bv, qb, kTb, vb);
}

// ===================== flash attention (1 wave / 16-query tile) =====================
__global__ __launch_bounds__(32) void k_attn(
    const unsigned short* __restrict__ qb, const unsigned short* __restrict__ kTb,
    const unsigned short* __restrict__ vb, const float* __restrict__ t,
    const float* __restrict__ time_decay, const unsigned char* __restrict__ mask,
    unsigned short* __restrict__ ob) {
  __shared__ unsigned short pbuf[16 * 32];
  int q0 = blockIdx.x * 16, h = blockIdx.y, b = blockIdx.z;
  int g = h / (H_ / G_);
  int lane = threadIdx.x, half = lane >> 4, nn = lane & 15;
  float decay = log1pf(__expf(time_decay[h]));
  const unsigned short* qbase = qb + ((long)(b * H_ + h) * S_ + q0) * HD_;
  v16bf aq0 = frag_a16(qbase, HD_);        // K = 0..31 of head dim
  v16bf aq1 = frag_a16(qbase + 32, HD_);   // K = 32..63
  float tq[8], mrow[8], lrow[8];
#pragma unroll
  for (int j = 0; j < 8; ++j) {
    tq[j] = t[b * S_ + q0 + j + half * 8];
    mrow[j] = -1e30f; lrow[j] = 0.f;
  }
  v8f oacc[4] = {};
  const unsigned short* kTh = kTb + (long)(b * G_ + g) * HD_ * S_;
  const unsigned short* vh  = vb  + (long)(b * G_ + g) * S_ * HD_;
  for (int kt = 0; kt < S_; kt += 32) {
    v8f s0 = {}, s1 = {};
    v16bf b00 = frag_b16(kTh + kt, S_);
    v16bf b01 = frag_b16(kTh + kt + 16, S_);
    v16bf b10 = frag_b16(kTh + (long)32 * S_ + kt, S_);
    v16bf b11 = frag_b16(kTh + (long)32 * S_ + kt + 16, S_);
    WMMA_BF16(s0, aq0, b00); WMMA_BF16(s0, aq1, b10);
    WMMA_BF16(s1, aq0, b01); WMMA_BF16(s1, aq1, b11);
    float tk0 = t[b * S_ + kt + nn], tk1 = t[b * S_ + kt + 16 + nn];
    bool mk0 = mask[b * S_ + kt + nn] != 0, mk1 = mask[b * S_ + kt + 16 + nn] != 0;
#pragma unroll
    for (int j = 0; j < 8; ++j) {
      float v0 = s0[j] * 0.125f - decay * fabsf(tq[j] - tk0);
      float v1 = s1[j] * 0.125f - decay * fabsf(tq[j] - tk1);
      if (!mk0) v0 = -1e9f;
      if (!mk1) v1 = -1e9f;
      float mx = fmaxf(v0, v1);
#pragma unroll
      for (int o = 1; o < 16; o <<= 1) mx = fmaxf(mx, __shfl_xor(mx, o, 32));
      float mnew = fmaxf(mrow[j], mx);
      float so = __expf(mrow[j] - mnew);
      float p0 = __expf(v0 - mnew), p1 = __expf(v1 - mnew);
      float ps = p0 + p1;
#pragma unroll
      for (int o = 1; o < 16; o <<= 1) ps += __shfl_xor(ps, o, 32);
      lrow[j] = lrow[j] * so + ps;
      mrow[j] = mnew;
#pragma unroll
      for (int c = 0; c < 4; ++c) oacc[c][j] *= so;
      pbuf[(j + half * 8) * 32 + nn]      = f2bf(p0);       // transpose P via LDS
      pbuf[(j + half * 8) * 32 + 16 + nn] = f2bf(p1);
    }
    __syncthreads();
    v16bf pa = frag_a16(pbuf, 32);
#pragma unroll
    for (int c = 0; c < 4; ++c) {
      v16bf bv_ = frag_b16(vh + (long)kt * HD_ + c * 16, HD_);
      WMMA_BF16(oacc[c], pa, bv_);
    }
    __syncthreads();
  }
#pragma unroll
  for (int j = 0; j < 8; ++j) {
    long tok = (long)b * S_ + q0 + j + half * 8;
    float inv = 1.f / lrow[j];
#pragma unroll
    for (int c = 0; c < 4; ++c)
      ob[tok * (H_ * HD_) + h * 64 + c * 16 + nn] = f2bf(oacc[c][j] * inv);
  }
}

// ===================== output projection + residual =====================
__global__ __launch_bounds__(256) void k_gemm_o(
    const unsigned short* __restrict__ A, const unsigned short* __restrict__ Wo,
    const float* __restrict__ xin, const float* __restrict__ bo, float* __restrict__ x2) {
  __shared__ unsigned short As[2][64][36];
  int m0 = blockIdx.x * 64, n0 = blockIdx.y * 64;
  int wave = threadIdx.x >> 5;
  int wm = wave & 3, wn = wave >> 2;
  int wn0 = n0 + wn * 32;
  v8f acc0 = {}, acc1 = {};
  auto stage = [&](int buf, int k0) {
    for (int i = threadIdx.x; i < 512; i += 256) {
      int r = i >> 3, c = (i & 7) << 2;
      async_g2l_b64(&As[buf][r][c], A + (long)(m0 + r) * D_ + k0 + c);
    }
  };
  stage(0, 0);
  wait_async0();
  __syncthreads();
  int buf = 0;
  for (int k0 = 0; k0 < D_; k0 += 32) {
    if (k0 + 32 < D_) stage(buf ^ 1, k0 + 32);
    __builtin_prefetch(Wo + (long)(k0 + 32) * D_ + wn0, 0, 1);
    v16bf af = frag_a16(&As[buf][wm * 16][0], 36);
    v16bf b0 = frag_b16(Wo + (long)k0 * D_ + wn0, D_);
    v16bf b1 = frag_b16(Wo + (long)k0 * D_ + wn0 + 16, D_);
    WMMA_BF16(acc0, af, b0);
    WMMA_BF16(acc1, af, b1);
    wait_async0();
    __syncthreads();
    buf ^= 1;
  }
  int lane = threadIdx.x & 31, half = lane >> 4;
#pragma unroll
  for (int p = 0; p < 2; ++p) {
    const v8f& acc = p ? acc1 : acc0;
    int n = wn0 + p * 16 + (lane & 15);
#pragma unroll
    for (int j = 0; j < 8; ++j) {
      long m = m0 + wm * 16 + j + half * 8;
      x2[m * D_ + n] = acc[j] + bo[n] + xin[m * D_ + n];
    }
  }
}

// ===================== LN2 + router (top-2) =====================
__global__ __launch_bounds__(256) void k_ln2_router(
    const float* __restrict__ x2, const float* __restrict__ sc, const float* __restrict__ bi,
    const float* __restrict__ Wr, const float* __restrict__ br,
    unsigned short* __restrict__ xn2, int* cnt, int* bucket, float* bgate) {
  __shared__ float sm[8];
  __shared__ float slog[E_];
  int tkn = blockIdx.x;
  long base = (long)tkn * D_;
  float s = 0.f;
  for (int d = threadIdx.x; d < D_; d += 256) s += x2[base + d];
  float mu = blockSum256(s, sm) * (1.f / D_);
  float vs = 0.f;
  for (int d = threadIdx.x; d < D_; d += 256) { float e = x2[base + d] - mu; vs += e * e; }
  float var = blockSum256(vs, sm) * (1.f / D_);
  float inv = rsqrtf(var + 1e-6f);
  if (threadIdx.x < E_) slog[threadIdx.x] = 0.f;
  __syncthreads();
  float lacc[E_];
#pragma unroll
  for (int e = 0; e < E_; ++e) lacc[e] = 0.f;
  for (int d = threadIdx.x; d < D_; d += 256) {
    float xn = (x2[base + d] - mu) * inv * sc[d] + bi[d];
    xn2[base + d] = f2bf(xn);
    const float* wr = Wr + (long)d * E_;
#pragma unroll
    for (int e = 0; e < E_; ++e) lacc[e] += xn * wr[e];
  }
#pragma unroll
  for (int e = 0; e < E_; ++e) atomicAdd(&slog[e], lacc[e]);   // LDS float atomics
  __syncthreads();
  if (threadIdx.x == 0) {
    float v[E_];
#pragma unroll
    for (int e = 0; e < E_; ++e) v[e] = slog[e] + br[e];
    int i1 = 0;
#pragma unroll
    for (int e = 1; e < E_; ++e) if (v[e] > v[i1]) i1 = e;
    int i2 = (i1 == 0) ? 1 : 0;
#pragma unroll
    for (int e = 0; e < E_; ++e) if (e != i1 && v[e] > v[i2]) i2 = e;
    float e2 = __expf(v[i2] - v[i1]);
    float g1 = 1.f / (1.f + e2), g2 = e2 / (1.f + e2);
    int p1 = atomicAdd(&cnt[i1], 1);
    bucket[i1 * NT + p1] = (tkn << 1);     bgate[i1 * NT + p1] = g1;
    int p2 = atomicAdd(&cnt[i2], 1);
    bucket[i2 * NT + p2] = (tkn << 1) | 1; bgate[i2 * NT + p2] = g2;
  }
}

// ===================== MoE expert GEMMs =====================
__global__ __launch_bounds__(256) void k_moe(
    const unsigned short* __restrict__ xn2, const unsigned short* __restrict__ W1b,
    const float* __restrict__ b1, const unsigned short* __restrict__ W2b,
    const float* __restrict__ b2, const int* __restrict__ cnt,
    const int* __restrict__ bucket, const float* __restrict__ bgate,
    float* __restrict__ yc) {
  int e = blockIdx.y;
  int t0 = blockIdx.x * 16;
  int nTok = cnt[e];
  if (t0 >= nTok) return;
  __shared__ unsigned short xs[16][D_];    // 32 KB gathered activations
  __shared__ unsigned short hs[16][128];   // 4 KB hidden chunk (post-GELU, bf16)
  __shared__ int scode[16];
  __shared__ float sgate[16];
  if (threadIdx.x < 16) {
    int idx = t0 + threadIdx.x;
    if (idx < nTok) { scode[threadIdx.x] = bucket[e * NT + idx]; sgate[threadIdx.x] = bgate[e * NT + idx]; }
    else            { scode[threadIdx.x] = -1;                   sgate[threadIdx.x] = 0.f; }
  }
  __syncthreads();
  // async gather of 16 token rows into LDS (padding rows load token 0; never stored)
  for (int i = threadIdx.x; i < 16 * 256; i += 256) {
    int r = i >> 8, c = (i & 255) << 2;
    int code = scode[r];
    long tok = (code >= 0) ? (long)(code >> 1) : 0;
    async_g2l_b64(&xs[r][c], xn2 + tok * D_ + c);
  }
  wait_async0();
  __syncthreads();
  int wave = threadIdx.x >> 5, lane = threadIdx.x & 31;
  int half = lane >> 4, nn = lane & 15;
  v8f acc[8] = {};
  const unsigned short* W1e = W1b + (long)e * D_ * FF_;
  const unsigned short* W2e = W2b + (long)e * FF_ * D_;
  int d0 = wave * 128;                       // this wave's 128 output columns
  for (int fc = 0; fc < FF_; fc += 128) {
    int f0 = fc + wave * 16;                 // this wave's 16 hidden columns
    v8f hacc = {};
    for (int k0 = 0; k0 < D_; k0 += 32) {
      __builtin_prefetch(W1e + (long)(k0 + 32) * FF_ + f0, 0, 1);
      v16bf a  = frag_a16(&xs[0][k0], D_);
      v16bf bw = frag_b16(W1e + (long)k0 * FF_ + f0, FF_);
      WMMA_BF16(hacc, a, bw);
    }
    float bias = b1[(long)e * FF_ + f0 + nn];
#pragma unroll
    for (int j = 0; j < 8; ++j) {
      float v = hacc[j] + bias;
      float u = v * 0.7978845608f * (1.f + 0.044715f * v * v);
      hs[j + half * 8][wave * 16 + nn] = f2bf(0.5f * v * (1.f + tanhf(u)));
    }
    __syncthreads();
#pragma unroll
    for (int ks = 0; ks < 4; ++ks) {
      v16bf a = frag_a16(&hs[0][ks * 32], 128);
#pragma unroll
      for (int c = 0; c < 8; ++c) {
        v16bf bw = frag_b16(W2e + (long)(fc + ks * 32) * D_ + d0 + c * 16, D_);
        WMMA_BF16(acc[c], a, bw);
      }
    }
    __syncthreads();
  }
#pragma unroll
  for (int c = 0; c < 8; ++c) {
    int col = d0 + c * 16 + nn;
    float bias2 = b2[(long)e * D_ + col];
#pragma unroll
    for (int j = 0; j < 8; ++j) {
      int r = j + half * 8;
      int code = scode[r];
      if (code >= 0)
        yc[(long)code * D_ + col] = sgate[r] * (acc[c][j] + bias2);  // slot = tok*2+rank
    }
  }
}

__global__ __launch_bounds__(256) void k_combine(const float* __restrict__ x2,
                                                 const float* __restrict__ yc,
                                                 float* __restrict__ out) {
  long tok = blockIdx.x;
  for (int d = threadIdx.x; d < D_; d += 256) {
    long i = tok * D_ + d;
    out[i] = x2[i] + yc[(tok * 2) * D_ + d] + yc[(tok * 2 + 1) * D_ + d];
  }
}

// ===================== host launcher =====================
extern "C" void kernel_launch(void* const* d_in, const int* in_sizes, int n_in,
                              void* d_out, int out_size, void* d_ws, size_t ws_size,
                              hipStream_t stream) {
  const float* x      = (const float*)d_in[0];
  const float* deltas = (const float*)d_in[1];
  const unsigned char* mask = (const unsigned char*)d_in[2];
  const float* ln1s = (const float*)d_in[3];
  const float* ln1b = (const float*)d_in[4];
  const float* ln2s = (const float*)d_in[5];
  const float* ln2b = (const float*)d_in[6];
  const float* Wq = (const float*)d_in[7];  const float* bq = (const float*)d_in[8];
  const float* Wk = (const float*)d_in[9];  const float* bk = (const float*)d_in[10];
  const float* Wv = (const float*)d_in[11]; const float* bv = (const float*)d_in[12];
  const float* Wo = (const float*)d_in[13]; const float* bo = (const float*)d_in[14];
  const float* tdec = (const float*)d_in[15];
  const float* Wr = (const float*)d_in[16]; const float* br = (const float*)d_in[17];
  const float* W1 = (const float*)d_in[18]; const float* b1 = (const float*)d_in[19];
  const float* W2 = (const float*)d_in[20]; const float* b2 = (const float*)d_in[21];
  float* out = (float*)d_out;

  char* w = (char*)d_ws;
  size_t off = 0;
  auto take = [&](size_t bytes) -> void* {
    void* p = w + off;
    off += (bytes + 255) & ~(size_t)255;
    return p;
  };
  float* t_f            = (float*)take(sizeof(float) * NT);
  unsigned short* Wqkvb = (unsigned short*)take(2ul * 1024 * 1536);
  unsigned short* xn1b  = (unsigned short*)take(2ul * NT * D_);
  unsigned short* qb    = (unsigned short*)take(2ul * B_ * H_ * S_ * HD_);
  unsigned short* kTb   = (unsigned short*)take(2ul * B_ * G_ * HD_ * S_);
  unsigned short* vb    = (unsigned short*)take(2ul * B_ * G_ * S_ * HD_);
  unsigned short* ob    = (unsigned short*)take(2ul * NT * D_);
  unsigned short* Wob   = (unsigned short*)take(2ul * 1024 * 1024);
  float* x2             = (float*)take(sizeof(float) * NT * D_);
  unsigned short* xn2b  = (unsigned short*)take(2ul * NT * D_);
  unsigned short* W1b   = (unsigned short*)take(2ul * E_ * D_ * FF_);
  unsigned short* W2b   = (unsigned short*)take(2ul * E_ * FF_ * D_);
  int*   cnt            = (int*)take(sizeof(int) * E_);
  int*   bucket         = (int*)take(sizeof(int) * E_ * NT);
  float* bgate          = (float*)take(sizeof(float) * E_ * NT);
  float* yc             = (float*)take(sizeof(float) * 2 * NT * D_);

  hipMemsetAsync(cnt, 0, sizeof(int) * E_, stream);
  k_cumsum<<<B_, 32, 0, stream>>>(deltas, t_f);

  k_convert<<<2048, 256, 0, stream>>>(Wq, Wqkvb, 1024, 1024, 1536, 0);
  k_convert<<<1024, 256, 0, stream>>>(Wk, Wqkvb, 1024, 256, 1536, 1024);
  k_convert<<<1024, 256, 0, stream>>>(Wv, Wqkvb, 1024, 256, 1536, 1280);
  k_convert<<<2048, 256, 0, stream>>>(Wo, Wob, 1024, 1024, 1024, 0);
  k_convert<<<4096, 256, 0, stream>>>(W1, W1b, E_ * D_, FF_, FF_, 0);
  k_convert<<<4096, 256, 0, stream>>>(W2, W2b, E_ * FF_, D_, D_, 0);

  k_ln1<<<NT, 256, 0, stream>>>(x, ln1s, ln1b, xn1b);
  k_gemm_qkv<<<dim3(64, 24), 256, 0, stream>>>(xn1b, Wqkvb, bq, bk, bv, qb, kTb, vb);
  k_attn<<<dim3(S_ / 16, H_, B_), 32, 0, stream>>>(qb, kTb, vb, t_f, tdec, mask, ob);
  k_gemm_o<<<dim3(64, 16), 256, 0, stream>>>(ob, Wob, x, bo, x2);
  k_ln2_router<<<NT, 256, 0, stream>>>(x2, ln2s, ln2b, Wr, br, xn2b, cnt, bucket, bgate);
  k_moe<<<dim3(NT / 16, E_), 256, 0, stream>>>(xn2b, W1b, b1, W2b, b2, cnt, bucket, bgate, yc);
  k_combine<<<NT, 256, 0, stream>>>(x2, yc, out);
}
